// DecoderWithAttention_42898133353401
// MI455X (gfx1250) — compile-verified, matrix operand order checked by reading, not run on hardware
//
#include <hip/hip_runtime.h>

#define B_    128
#define R_    196
#define ENC_  2048
#define ATT_  512
#define DEC_  512
#define EMB_  512
#define V_    10000
#define T_    20
#define XDIM_ 3072   /* EMB + ENC + DEC packed activation row */

typedef __attribute__((ext_vector_type(16))) _Float16 v16h;
typedef __attribute__((ext_vector_type(8)))  _Float16 v8h;
typedef __attribute__((ext_vector_type(8)))  float    v8f;
typedef __attribute__((ext_vector_type(4)))  float    v4f;

__device__ __forceinline__ float sigmoidf_(float x) { return 1.0f / (1.0f + __expf(-x)); }

// ---- WMMA fragment loads (16-bit 16x32 A layout, ISA 7.12.2) ----
// lane sub = lane&15 selects row; hi = lane>>4 selects K subgroup:
//   halfs[0..7]  = K (k0 + hi*8) .. +7 ; halfs[8..15] = K (k0 + hi*8 + 16) .. +23
__device__ __forceinline__ v16h load_frag_h16(const _Float16* p) {
  v8h lo = *(const v8h*)p;
  v8h hi = *(const v8h*)(p + 16);
  v16h r;
#pragma unroll
  for (int i = 0; i < 8; ++i) { r[i] = lo[i]; r[i + 8] = hi[i]; }
  return r;
}

__device__ __forceinline__ v16h load_frag_f32(const float* p) {
  v4f a0 = *(const v4f*)(p);
  v4f a1 = *(const v4f*)(p + 4);
  v4f b0 = *(const v4f*)(p + 16);
  v4f b1 = *(const v4f*)(p + 20);
  v16h r;
#pragma unroll
  for (int i = 0; i < 4; ++i) {
    r[i]      = (_Float16)a0[i];
    r[i + 4]  = (_Float16)a1[i];
    r[i + 8]  = (_Float16)b0[i];
    r[i + 12] = (_Float16)b1[i];
  }
  return r;
}

// ---- K-loop body; CLAMP path only taken by the single ragged N block ----
// All 4 B fragments (and A) are loaded into distinct registers BEFORE the
// WMMA chain so the backend can emit one load clause and staggered partial
// s_wait_loadcnt values instead of draining loads before every WMMA.
template<bool AF32, bool CLAMP>
__device__ __forceinline__ void gemm_kloop(
    const void* __restrict__ Av, long arow,
    const _Float16* __restrict__ W, int ldw,
    int n0, int sub, int hi, int N, int K, v8f acc[4])
{
  int wr[4];
#pragma unroll
  for (int t = 0; t < 4; ++t) {
    wr[t] = n0 + t * 16 + sub;
    if (CLAMP && wr[t] >= N) wr[t] = N - 1;    // clamp load; masked on store
  }
  for (int k0 = 0; k0 < K; k0 += 32) {
    v16h a;
    if (AF32) a = load_frag_f32((const float*)Av + arow + k0 + hi * 8);
    else      a = load_frag_h16((const _Float16*)Av + arow + k0 + hi * 8);
    v16h b0 = load_frag_h16(W + (long)wr[0] * ldw + k0 + hi * 8);
    v16h b1 = load_frag_h16(W + (long)wr[1] * ldw + k0 + hi * 8);
    v16h b2 = load_frag_h16(W + (long)wr[2] * ldw + k0 + hi * 8);
    v16h b3 = load_frag_h16(W + (long)wr[3] * ldw + k0 + hi * 8);
    acc[0] = __builtin_amdgcn_wmma_f32_16x16x32_f16(false, a, false, b0, (short)0, acc[0], false, false);
    acc[1] = __builtin_amdgcn_wmma_f32_16x16x32_f16(false, a, false, b1, (short)0, acc[1], false, false);
    acc[2] = __builtin_amdgcn_wmma_f32_16x16x32_f16(false, a, false, b2, (short)0, acc[2], false, false);
    acc[3] = __builtin_amdgcn_wmma_f32_16x16x32_f16(false, a, false, b3, (short)0, acc[3], false, false);
  }
}

// ---- Generic WMMA GEMM: C[M,N] = A[M,K] * W[N,K]^T + bias[N] ----
// M % 128 == 0, K % 32 == 0. 256 threads = 8 waves; wave w owns rows
// blockIdx.y*128 + w*16; block owns 64 columns. EPI: 0=f32, 1=sigmoid f32,
// 2=f16 store. AF32: A operand is f32 (converted in-register).
template<bool AF32, int EPI>
__global__ void __launch_bounds__(256) gemm_wmma_kernel(
    const void* __restrict__ Av, int lda,
    const _Float16* __restrict__ W, int ldw,
    const float* __restrict__ bias,
    void* __restrict__ Cv, long ldc, int N, int K)
{
  const int lane = threadIdx.x & 31;
  const int wave = threadIdx.x >> 5;
  const int sub  = lane & 15;
  const int hi   = lane >> 4;
  const int m0   = blockIdx.y * 128 + wave * 16;
  const int n0   = blockIdx.x * 64;

  v8f acc[4] = {};
  const long arow = (long)(m0 + sub) * lda;

  if (n0 + 64 <= N)  // uniform branch; EXEC stays all-ones around WMMA
    gemm_kloop<AF32, false>(Av, arow, W, ldw, n0, sub, hi, N, K, acc);
  else
    gemm_kloop<AF32, true >(Av, arow, W, ldw, n0, sub, hi, N, K, acc);

#pragma unroll
  for (int t = 0; t < 4; ++t) {
    const int col = n0 + t * 16 + sub;
    if (col >= N) continue;
    const float bv = bias[col];
#pragma unroll
    for (int v = 0; v < 8; ++v) {
      const long row = m0 + v + hi * 8;        // C layout: VGPR v -> M=v(+8)
      float val = acc[t][v] + bv;
      if (EPI == 1) val = sigmoidf_(val);
      if (EPI == 2) ((_Float16*)Cv)[row * ldc + col] = (_Float16)val;
      else          ((float*)Cv)[row * ldc + col] = val;
    }
  }
}

// ---- score = relu(att1 + att2) . w_full ; softmax over R -> alpha ----
__global__ void __launch_bounds__(256) score_softmax_kernel(
    const _Float16* __restrict__ att1, const float* __restrict__ att2,
    const float* __restrict__ wfull, float* __restrict__ alpha,
    float* __restrict__ alphas_out, int t)
{
  const int b = blockIdx.x;
  __shared__ float s_att2[ATT_];
  __shared__ float s_wf[ATT_];
  __shared__ float s_score[R_];
  __shared__ float s_red[8];

  for (int i = threadIdx.x; i < ATT_; i += 256) {
    s_att2[i] = att2[b * ATT_ + i];
    s_wf[i]   = wfull[i];
  }
  __syncthreads();

  const int wave = threadIdx.x >> 5, lane = threadIdx.x & 31;
  for (int r = wave; r < R_; r += 8) {
    const _Float16* row = att1 + ((long)b * R_ + r) * ATT_;
    float sum = 0.f;
    for (int k = lane; k < ATT_; k += 32) {
      float v = (float)row[k] + s_att2[k];
      v = v > 0.f ? v : 0.f;
      sum += v * s_wf[k];
    }
#pragma unroll
    for (int off = 16; off > 0; off >>= 1) sum += __shfl_xor(sum, off);
    if (lane == 0) s_score[r] = sum;
  }
  __syncthreads();

  float v = (threadIdx.x < R_) ? s_score[threadIdx.x] : -3.0e38f;
  float m = v;
#pragma unroll
  for (int off = 16; off > 0; off >>= 1) m = fmaxf(m, __shfl_xor(m, off));
  if (lane == 0) s_red[wave] = m;
  __syncthreads();
  float mx = fmaxf(fmaxf(fmaxf(s_red[0], s_red[1]), fmaxf(s_red[2], s_red[3])),
                   fmaxf(fmaxf(s_red[4], s_red[5]), fmaxf(s_red[6], s_red[7])));
  __syncthreads();

  float e = (threadIdx.x < R_) ? __expf(v - mx) : 0.f;
  float s = e;
#pragma unroll
  for (int off = 16; off > 0; off >>= 1) s += __shfl_xor(s, off);
  if (lane == 0) s_red[wave] = s;
  __syncthreads();
  float tot = s_red[0] + s_red[1] + s_red[2] + s_red[3] +
              s_red[4] + s_red[5] + s_red[6] + s_red[7];
  if (threadIdx.x < R_) {
    float a = e / tot;
    alpha[b * R_ + threadIdx.x] = a;
    alphas_out[((long)b * T_ + t) * R_ + threadIdx.x] = a;
  }
}

// ---- context = gate * (alpha . encoder_out), pack x = [emb_t, context, h] ----
// ET = _Float16 (preferred, halves the dominant 4.1 GB streaming term) or float.
template<typename ET>
__global__ void __launch_bounds__(256) context_pack_kernel(
    const ET* __restrict__ enc, const float* __restrict__ alpha,
    const float* __restrict__ gate, const _Float16* __restrict__ emb,
    _Float16* __restrict__ x_ext, int t)
{
  const int b = blockIdx.x;
  __shared__ float s_alpha[R_];
  for (int i = threadIdx.x; i < R_; i += 256) s_alpha[i] = alpha[b * R_ + i];
  __syncthreads();

  float acc[8] = {0.f, 0.f, 0.f, 0.f, 0.f, 0.f, 0.f, 0.f};
  const ET* eb = enc + (long)b * R_ * ENC_;
  for (int r = 0; r < R_; ++r) {
    if (r + 8 < R_)  // hide HBM/L2 latency on the streaming operand
      __builtin_prefetch(eb + (long)(r + 8) * ENC_ + (threadIdx.x << 3), 0, 3);
    const float a = s_alpha[r];
    const ET* er = eb + (long)r * ENC_;
#pragma unroll
    for (int j = 0; j < 8; ++j) acc[j] += a * (float)er[threadIdx.x + j * 256];
  }
  _Float16* xb = x_ext + (long)b * XDIM_;
#pragma unroll
  for (int j = 0; j < 8; ++j) {
    const int e = threadIdx.x + j * 256;
    xb[EMB_ + e] = (_Float16)(gate[b * ENC_ + e] * acc[j]);
  }
  const _Float16* et = emb + ((long)b * T_ + t) * EMB_;
  for (int i = threadIdx.x; i < EMB_; i += 256) xb[i] = et[i];
}

// ---- LSTM pointwise update; writes h (f16) into packed x row ----
__global__ void __launch_bounds__(256) lstm_update_kernel(
    const float* __restrict__ gates, float* __restrict__ c,
    _Float16* __restrict__ x_ext)
{
  const int idx = blockIdx.x * 256 + threadIdx.x;  // B_*DEC_ = 65536
  const int b = idx >> 9, d = idx & 511;
  const float* g = gates + (long)b * (4 * DEC_);
  const float i_ = sigmoidf_(g[d]);
  const float f_ = sigmoidf_(g[DEC_ + d]);
  const float gg = tanhf(g[2 * DEC_ + d]);
  const float o_ = sigmoidf_(g[3 * DEC_ + d]);
  const float cn = f_ * c[idx] + i_ * gg;
  c[idx] = cn;
  x_ext[(long)b * XDIM_ + EMB_ + ENC_ + d] = (_Float16)(o_ * tanhf(cn));
}

// ---- mean over R ----
__global__ void __launch_bounds__(256) mean_enc_kernel(
    const float* __restrict__ enc, float* __restrict__ mean)
{
  const int b = blockIdx.x;
  const float* eb = enc + (long)b * R_ * ENC_;
  float acc[8] = {0.f, 0.f, 0.f, 0.f, 0.f, 0.f, 0.f, 0.f};
  for (int r = 0; r < R_; ++r) {
    const float* er = eb + (long)r * ENC_;
#pragma unroll
    for (int j = 0; j < 8; ++j) acc[j] += er[threadIdx.x + j * 256];
  }
#pragma unroll
  for (int j = 0; j < 8; ++j)
    mean[b * ENC_ + threadIdx.x + j * 256] = acc[j] * (1.0f / (float)R_);
}

// ---- misc small kernels ----
__global__ void f32_to_f16_kernel(const float* __restrict__ in,
                                  _Float16* __restrict__ out, long n) {
  long i = (long)blockIdx.x * 256 + threadIdx.x;
  if (i < n) out[i] = (_Float16)in[i];
}

__global__ void pack_wcat_kernel(const float* __restrict__ wih,
                                 const float* __restrict__ whh,
                                 _Float16* __restrict__ wcat) {
  long i = (long)blockIdx.x * 256 + threadIdx.x;   // 2048 * 3072
  if (i >= (long)2048 * XDIM_) return;
  const int n = (int)(i / XDIM_), k = (int)(i % XDIM_);
  float v = (k < EMB_ + ENC_) ? wih[(long)n * (EMB_ + ENC_) + k]
                              : whh[(long)n * DEC_ + (k - EMB_ - ENC_)];
  wcat[i] = (_Float16)v;
}

__global__ void bias_sum_kernel(const float* __restrict__ a,
                                const float* __restrict__ b,
                                float* __restrict__ out) {
  int i = blockIdx.x * 256 + threadIdx.x;
  if (i < 2048) out[i] = a[i] + b[i];
}

__global__ void embed_gather_kernel(const int* __restrict__ captions,
                                    const float* __restrict__ table,
                                    _Float16* __restrict__ emb) {
  long i = (long)blockIdx.x * 256 + threadIdx.x;   // B*T*EMB
  if (i >= (long)B_ * T_ * EMB_) return;
  const int e = (int)(i & 511);
  const long bt = i >> 9;
  const int t = (int)(bt % T_), b = (int)(bt / T_);
  const int tok = captions[b * (T_ + 1) + t];
  emb[i] = (_Float16)table[(long)tok * EMB_ + e];
}

extern "C" void kernel_launch(void* const* d_in, const int* in_sizes, int n_in,
                              void* d_out, int out_size, void* d_ws, size_t ws_size,
                              hipStream_t stream)
{
  const float* encoder_out = (const float*)d_in[0];
  const int*   captions    = (const int*)d_in[1];
  const float* W_enc_att   = (const float*)d_in[2];
  const float* b_enc_att   = (const float*)d_in[3];
  const float* W_dec_att   = (const float*)d_in[4];
  const float* b_dec_att   = (const float*)d_in[5];
  const float* W_full_att  = (const float*)d_in[6];
  /* b_full_att (d_in[7]) is a constant added to every score: softmax-invariant */
  const float* emb_table   = (const float*)d_in[8];
  const float* W_ih        = (const float*)d_in[9];
  const float* W_hh        = (const float*)d_in[10];
  const float* b_ih        = (const float*)d_in[11];
  const float* b_hh        = (const float*)d_in[12];
  const float* W_init_h    = (const float*)d_in[13];
  const float* b_init_h    = (const float*)d_in[14];
  const float* W_init_c    = (const float*)d_in[15];
  const float* b_init_c    = (const float*)d_in[16];
  const float* W_fbeta     = (const float*)d_in[17];
  const float* b_fbeta     = (const float*)d_in[18];
  const float* W_fc        = (const float*)d_in[19];
  const float* b_fc        = (const float*)d_in[20];

  float* preds_out  = (float*)d_out;                           // (B,T,V)
  float* alphas_out = (float*)d_out + (long)B_ * T_ * V_;      // (B,T,R)

  size_t off = 0;
  auto alloc = [&](size_t bytes) -> void* {
    void* p = (char*)d_ws + off;
    off += (bytes + 255) & ~(size_t)255;
    return p;
  };
  _Float16* att1_f16   = (_Float16*)alloc((size_t)B_ * R_ * ATT_ * 2);
  _Float16* Wenc_f16   = (_Float16*)alloc((size_t)ATT_ * ENC_ * 2);
  _Float16* Wdec_f16   = (_Float16*)alloc((size_t)ATT_ * DEC_ * 2);
  _Float16* Wcat_f16   = (_Float16*)alloc((size_t)2048 * XDIM_ * 2);
  _Float16* Wfc_f16    = (_Float16*)alloc((size_t)V_ * DEC_ * 2);
  _Float16* Wfb_f16    = (_Float16*)alloc((size_t)ENC_ * DEC_ * 2);
  _Float16* Wih0_f16   = (_Float16*)alloc((size_t)DEC_ * ENC_ * 2);  // W_init_h
  _Float16* Wic0_f16   = (_Float16*)alloc((size_t)DEC_ * ENC_ * 2);  // W_init_c
  _Float16* emb_f16    = (_Float16*)alloc((size_t)B_ * T_ * EMB_ * 2);
  _Float16* x_ext      = (_Float16*)alloc((size_t)B_ * XDIM_ * 2);
  float*    mean_f32   = (float*)alloc((size_t)B_ * ENC_ * 4);
  float*    att2_f32   = (float*)alloc((size_t)B_ * ATT_ * 4);
  float*    alpha_f32  = (float*)alloc((size_t)B_ * R_ * 4);
  float*    gate_f32   = (float*)alloc((size_t)B_ * ENC_ * 4);
  float*    gates_f32  = (float*)alloc((size_t)B_ * 2048 * 4);
  float*    c_state    = (float*)alloc((size_t)B_ * DEC_ * 4);
  float*    bsum       = (float*)alloc((size_t)2048 * 4);

  // Optional f16 copy of encoder_out: halves the dominant 20x context-stream
  // traffic and removes cvt ops from the att1 GEMM A-path. Allocated last;
  // falls back to the f32 paths if the workspace is too small (decision is a
  // pure function of ws_size -> deterministic across calls).
  _Float16* enc_f16 = nullptr;
  {
    const size_t need = ((size_t)B_ * R_ * ENC_ * 2 + 255) & ~(size_t)255;
    if (off + need <= ws_size) enc_f16 = (_Float16*)alloc((size_t)B_ * R_ * ENC_ * 2);
  }
  const bool use16 = (enc_f16 != nullptr);
  (void)n_in; (void)in_sizes; (void)out_size;

  auto cvt = [&](const float* src, _Float16* dst, long n) {
    f32_to_f16_kernel<<<dim3((unsigned)((n + 255) / 256)), 256, 0, stream>>>(src, dst, n);
  };

  // ---- one-time weight conversion / packing ----
  cvt(W_enc_att, Wenc_f16, (long)ATT_ * ENC_);
  cvt(W_dec_att, Wdec_f16, (long)ATT_ * DEC_);
  cvt(W_fbeta,   Wfb_f16,  (long)ENC_ * DEC_);
  cvt(W_fc,      Wfc_f16,  (long)V_ * DEC_);
  cvt(W_init_h,  Wih0_f16, (long)DEC_ * ENC_);
  cvt(W_init_c,  Wic0_f16, (long)DEC_ * ENC_);
  if (use16) cvt(encoder_out, enc_f16, (long)B_ * R_ * ENC_);
  pack_wcat_kernel<<<dim3((unsigned)(((long)2048 * XDIM_ + 255) / 256)), 256, 0, stream>>>(
      W_ih, W_hh, Wcat_f16);
  bias_sum_kernel<<<dim3(8), 256, 0, stream>>>(b_ih, b_hh, bsum);
  embed_gather_kernel<<<dim3((unsigned)(((long)B_ * T_ * EMB_ + 255) / 256)), 256, 0, stream>>>(
      captions, emb_table, emb_f16);
  mean_enc_kernel<<<dim3(B_), 256, 0, stream>>>(encoder_out, mean_f32);

  // ---- att1 = encoder_out @ W_enc_att^T + b_enc_att  (f16 out) ----
  if (use16)
    gemm_wmma_kernel<false, 2><<<dim3(ATT_ / 64, (B_ * R_) / 128), 256, 0, stream>>>(
        enc_f16, ENC_, Wenc_f16, ENC_, b_enc_att, att1_f16, (long)ATT_, ATT_, ENC_);
  else
    gemm_wmma_kernel<true, 2><<<dim3(ATT_ / 64, (B_ * R_) / 128), 256, 0, stream>>>(
        encoder_out, ENC_, Wenc_f16, ENC_, b_enc_att, att1_f16, (long)ATT_, ATT_, ENC_);

  // ---- h0 (f16, packed into x_ext h-slot) and c0 (f32) ----
  gemm_wmma_kernel<true, 2><<<dim3(DEC_ / 64, 1), 256, 0, stream>>>(
      mean_f32, ENC_, Wih0_f16, ENC_, b_init_h, x_ext + EMB_ + ENC_, (long)XDIM_, DEC_, ENC_);
  gemm_wmma_kernel<true, 0><<<dim3(DEC_ / 64, 1), 256, 0, stream>>>(
      mean_f32, ENC_, Wic0_f16, ENC_, b_init_c, c_state, (long)DEC_, DEC_, ENC_);

  // ---- sequential recurrence ----
  for (int t = 0; t < T_; ++t) {
    // att2 = h @ W_dec_att^T + b_dec_att
    gemm_wmma_kernel<false, 0><<<dim3(ATT_ / 64, 1), 256, 0, stream>>>(
        x_ext + EMB_ + ENC_, XDIM_, Wdec_f16, DEC_, b_dec_att, att2_f32, (long)ATT_, ATT_, DEC_);
    // alpha = softmax_r( relu(att1+att2).w_full )
    score_softmax_kernel<<<dim3(B_), 256, 0, stream>>>(
        att1_f16, att2_f32, W_full_att, alpha_f32, alphas_out, t);
    // gate = sigmoid(h @ W_fbeta^T + b_fbeta)
    gemm_wmma_kernel<false, 1><<<dim3(ENC_ / 64, 1), 256, 0, stream>>>(
        x_ext + EMB_ + ENC_, XDIM_, Wfb_f16, DEC_, b_fbeta, gate_f32, (long)ENC_, ENC_, DEC_);
    // context = gate * (alpha . enc); pack [emb_t, context] into x row
    if (use16)
      context_pack_kernel<_Float16><<<dim3(B_), 256, 0, stream>>>(
          enc_f16, alpha_f32, gate_f32, emb_f16, x_ext, t);
    else
      context_pack_kernel<float><<<dim3(B_), 256, 0, stream>>>(
          encoder_out, alpha_f32, gate_f32, emb_f16, x_ext, t);
    // gates = x_ext @ [W_ih|W_hh]^T + (b_ih + b_hh)
    gemm_wmma_kernel<false, 0><<<dim3(2048 / 64, 1), 256, 0, stream>>>(
        x_ext, XDIM_, Wcat_f16, XDIM_, bsum, gates_f32, (long)2048, 2048, XDIM_);
    // LSTM pointwise; h (f16) back into x_ext
    lstm_update_kernel<<<dim3((B_ * DEC_) / 256), 256, 0, stream>>>(
        gates_f32, c_state, x_ext);
    // pred[:, t, :] = h @ W_fc^T + b_fc
    gemm_wmma_kernel<false, 0><<<dim3((V_ + 63) / 64, 1), 256, 0, stream>>>(
        x_ext + EMB_ + ENC_, XDIM_, Wfc_f16, DEC_, b_fc,
        preds_out + (long)t * V_, (long)T_ * V_, V_, DEC_);
  }
}
